// Model3DETR_51488067944657
// MI455X (gfx1250) — matrix operand order repaired and checked by reading, not compiled, not run on hardware
//
#include <hip/hip_runtime.h>

#define BATCH 4
#define NPTS 16000
#define NPRE 2048
#define NQc 256
#define DM 256
#define NHh 4
#define HDIM 64
#define NSAMP 32
#define NENCL 3
#define NDECL 8
#define NSEM 18
#define NANG 12
#define PI_F 3.14159265358979323846f

typedef unsigned int u32;
typedef unsigned short u16;
typedef __attribute__((ext_vector_type(16))) __bf16 v16bf;
typedef __attribute__((ext_vector_type(8))) float v8f;
typedef __attribute__((ext_vector_type(4))) int v4i;

union AccF { v8f v; float f[8]; };
union FragB { v16bf v; u32 u[8]; };

__device__ __forceinline__ u16 f2bf(float x) {
  u32 u = __float_as_uint(x);
  u32 r = u + 0x7fffu + ((u >> 16) & 1u);
  return (u16)(r >> 16);
}
__device__ __forceinline__ float sigm(float x) { return 1.0f / (1.0f + __expf(-x)); }

// ---- gfx1250 async global->LDS staging (guarded; falls back to sync copies)
#if defined(__has_builtin)
#if __has_builtin(__builtin_amdgcn_global_load_async_to_lds_b128)
#define HAS_ASYNC_LDS 1
#endif
#endif

typedef __attribute__((address_space(1))) v4i* gptr4;
typedef __attribute__((address_space(3))) v4i* lptr4;

__device__ __forceinline__ void cp16(const u16* g, u16* l) {
#ifdef HAS_ASYNC_LDS
  __builtin_amdgcn_global_load_async_to_lds_b128((gptr4)(v4i*)g, (lptr4)(v4i*)l, 0, 0);
#else
  const u32* gs = (const u32*)g;
  u32* ls = (u32*)l;
  ls[0] = gs[0]; ls[1] = gs[1]; ls[2] = gs[2]; ls[3] = gs[3];
#endif
}
__device__ __forceinline__ void async_wait() {
#ifdef HAS_ASYNC_LDS
#if __has_builtin(__builtin_amdgcn_s_wait_asynccnt)
  __builtin_amdgcn_s_wait_asynccnt(0);
#else
  asm volatile("s_wait_asynccnt 0" ::: "memory");
#endif
#endif
}

// ---------------------------------------------------------------- f32 -> bf16 converters
__global__ void cvt_bf16_kernel(const float* __restrict__ src, u16* __restrict__ dst, int n) {
  int t = blockIdx.x * blockDim.x + threadIdx.x;
  if (t < n) dst[t] = f2bf(src[t]);
}
// W (K x N f32) -> Wt (N x K bf16)
__global__ void cvt_w_t_kernel(const float* __restrict__ W, u16* __restrict__ Wt) {
  int t = blockIdx.x * blockDim.x + threadIdx.x;  // 65536
  int n = t >> 8, k = t & 255;
  Wt[t] = f2bf(W[k * DM + n]);
}
// V (B,Tk,NH,64 f32) -> Vt (B,NH,64,Tk bf16)
__global__ void cvt_v_t_kernel(const float* __restrict__ V, u16* __restrict__ Vt, int Tk) {
  int t = blockIdx.x * blockDim.x + threadIdx.x;
  if (t >= BATCH * NHh * HDIM * Tk) return;
  int j = t % Tk;
  int d = (t / Tk) & 63;
  int h = (t / (Tk * HDIM)) % NHh;
  int b = t / (Tk * HDIM * NHh);
  Vt[t] = f2bf(V[((size_t)b * Tk + j) * DM + h * HDIM + d]);
}

// ---------------------------------------------------------------- min/max
__global__ void minmax_kernel(const float* __restrict__ pc, float* pcmin, float* pcmax) {
  __shared__ float smn[256][3], smx[256][3];
  int b = blockIdx.x, t = threadIdx.x;
  float mn[3] = {1e30f, 1e30f, 1e30f}, mx[3] = {-1e30f, -1e30f, -1e30f};
  for (int i = t; i < NPTS; i += 256) {
    const float* p = pc + ((size_t)b * NPTS + i) * 3;
    for (int c = 0; c < 3; ++c) { mn[c] = fminf(mn[c], p[c]); mx[c] = fmaxf(mx[c], p[c]); }
  }
  for (int c = 0; c < 3; ++c) { smn[t][c] = mn[c]; smx[t][c] = mx[c]; }
  __syncthreads();
  for (int off = 128; off > 0; off >>= 1) {
    if (t < off)
      for (int c = 0; c < 3; ++c) {
        smn[t][c] = fminf(smn[t][c], smn[t + off][c]);
        smx[t][c] = fmaxf(smx[t][c], smx[t + off][c]);
      }
    __syncthreads();
  }
  if (t == 0)
    for (int c = 0; c < 3; ++c) { pcmin[b * 3 + c] = smn[0][c]; pcmax[b * 3 + c] = smx[0][c]; }
}

// ---------------------------------------------------------------- FPS
__global__ __launch_bounds__(512) void fps_kernel(const float* __restrict__ xyz, int* __restrict__ inds,
                                                  float* __restrict__ dist, int n, int npoint) {
  __shared__ float rv[512];
  __shared__ int ra[512];
  __shared__ int sfar;
  int b = blockIdx.x, t = threadIdx.x;
  const float* X = xyz + (size_t)b * n * 3;
  float* di = dist + (size_t)b * n;
  for (int i = t; i < n; i += 512) di[i] = 1e10f;
  __syncthreads();
  int far = 0;
  for (int s = 0; s < npoint; ++s) {
    if (t == 0) inds[b * npoint + s] = far;
    float cx = X[far * 3 + 0], cy = X[far * 3 + 1], cz = X[far * 3 + 2];
    float bm = -1.f; int ba = 0;
    for (int i = t; i < n; i += 512) {
      float dx = X[i * 3 + 0] - cx, dy = X[i * 3 + 1] - cy, dz = X[i * 3 + 2] - cz;
      float d = dx * dx + dy * dy + dz * dz;
      float nd = fminf(di[i], d);
      di[i] = nd;
      if (nd > bm) { bm = nd; ba = i; }  // ascending i => lowest-index tie-break
    }
    rv[t] = bm; ra[t] = ba;
    __syncthreads();
    for (int off = 256; off > 0; off >>= 1) {
      if (t < off) {
        if (rv[t + off] > rv[t] || (rv[t + off] == rv[t] && ra[t + off] < ra[t])) {
          rv[t] = rv[t + off]; ra[t] = ra[t + off];
        }
      }
      __syncthreads();
    }
    if (t == 0) sfar = ra[0];
    __syncthreads();
    far = sfar;
  }
}

// ---------------------------------------------------------------- gather rows of xyz
__global__ void gather3_kernel(const float* __restrict__ src, const int* __restrict__ inds,
                               float* __restrict__ dst, int n, int m) {
  int t = blockIdx.x * blockDim.x + threadIdx.x;
  if (t >= BATCH * m) return;
  int b = t / m, i = t % m;
  int id = inds[b * m + i];
  for (int c = 0; c < 3; ++c)
    dst[((size_t)b * m + i) * 3 + c] = src[((size_t)b * n + id) * 3 + c];
}

// ---------------------------------------------------------------- ball query (lax.top_k semantics)
__global__ void ballq_kernel(const float* __restrict__ xyz, const float* __restrict__ ctr,
                             int* __restrict__ out, int n, int m, float r2) {
  int t = blockIdx.x * blockDim.x + threadIdx.x;
  if (t >= BATCH * m) return;
  int b = t / m, q = t % m;
  const float* X = xyz + (size_t)b * n * 3;
  const float* C = ctr + ((size_t)b * m + q) * 3;
  float ax = C[0], ay = C[1], az = C[2];
  int* o = out + ((size_t)b * m + q) * NSAMP;
  int cnt = 0, first = 0;
  for (int i = 0; i < n; ++i) {
    float dx = X[i * 3] - ax, dy = X[i * 3 + 1] - ay, dz = X[i * 3 + 2] - az;
    float d2 = dx * dx + dy * dy + dz * dz;
    if (d2 < r2) {
      if (cnt == 0) first = i;
      o[cnt++] = i;
      if (cnt == NSAMP) break;
    }
  }
  for (; cnt < NSAMP; ++cnt) o[cnt] = first;
}

// ---------------------------------------------------------------- pre-MLP (3->64->128->256, max over 32 samples)
__global__ __launch_bounds__(256) void premlp_kernel(
    const float* __restrict__ xyz, const float* __restrict__ pre_xyz, const int* __restrict__ nidx,
    const float* __restrict__ w1, const float* __restrict__ b1,
    const float* __restrict__ w2, const float* __restrict__ b2,
    const float* __restrict__ w3, const float* __restrict__ b3,
    float* __restrict__ out) {
  __shared__ float g[NSAMP][3];
  __shared__ float h1[NSAMP][64];
  __shared__ float h2[NSAMP][128];
  int b = blockIdx.x / NPRE, p = blockIdx.x % NPRE, t = threadIdx.x;
  if (t < NSAMP * 3) {
    int s = t / 3, c = t % 3;
    int id = nidx[((size_t)b * NPRE + p) * NSAMP + s];
    g[s][c] = xyz[((size_t)b * NPTS + id) * 3 + c] - pre_xyz[((size_t)b * NPRE + p) * 3 + c];
  }
  __syncthreads();
  for (int idx = t; idx < NSAMP * 64; idx += 256) {
    int s = idx >> 6, j = idx & 63;
    float a = b1[j];
    for (int c = 0; c < 3; ++c) a += g[s][c] * w1[c * 64 + j];
    h1[s][j] = fmaxf(a, 0.f);
  }
  __syncthreads();
  for (int idx = t; idx < NSAMP * 128; idx += 256) {
    int s = idx >> 7, j = idx & 127;
    float a = b2[j];
    for (int c = 0; c < 64; ++c) a += h1[s][c] * w2[c * 128 + j];
    h2[s][j] = fmaxf(a, 0.f);
  }
  __syncthreads();
  int j = t;
  float m = -1e30f;
  for (int s = 0; s < NSAMP; ++s) {
    float a = b3[j];
    for (int c = 0; c < 128; ++c) a += h2[s][c] * w3[c * 256 + j];
    m = fmaxf(m, fmaxf(a, 0.f));
  }
  out[((size_t)b * NPRE + p) * DM + j] = m;
}

// ---------------------------------------------------------------- WMMA GEMM on pre-converted bf16
// C[M,N] = act(A_bf[M,K] * Wt_bf[N,K]^T + bias); tiles staged async into LDS.
__global__ __launch_bounds__(256) void gemm_bf_kernel(
    const u16* __restrict__ A, const u16* __restrict__ Wt, const float* __restrict__ bias,
    float* __restrict__ C, int M, int N, int K, int relu) {
  __shared__ __align__(16) u16 As[64][40];  // [m][k], 80B row stride (16B multiple)
  __shared__ __align__(16) u16 Ws[64][40];  // [n][k]
  const int tid = threadIdx.x;
  const int lane = tid & 31, wid = tid >> 5;
  const int hl = lane >> 4, lm = lane & 15;
  const int m_off = (wid & 3) * 16, n_off = (wid >> 2) * 32;
  const int bm = blockIdx.y * 64, bn = blockIdx.x * 64;
  const int r = tid >> 2, c4 = tid & 3;  // one 16B chunk per thread per matrix

  AccF c0, c1;
  for (int i = 0; i < 8; ++i) { c0.f[i] = 0.f; c1.f[i] = 0.f; }

  for (int k0 = 0; k0 < K; k0 += 32) {
    cp16(A + (size_t)(bm + r) * K + k0 + c4 * 8, &As[r][c4 * 8]);
    cp16(Wt + (size_t)(bn + r) * K + k0 + c4 * 8, &Ws[r][c4 * 8]);
    if (k0 + 32 < K) {  // prefetch next tiles (global_prefetch_b8)
      __builtin_prefetch(A + (size_t)(bm + r) * K + k0 + 32 + c4 * 8, 0, 1);
      __builtin_prefetch(Wt + (size_t)(bn + r) * K + k0 + 32 + c4 * 8, 0, 1);
    }
    async_wait();
    __syncthreads();
    FragB af, b0, b1;
    const u32* ar = (const u32*)&As[m_off + lm][0];
    const u32* br0 = (const u32*)&Ws[n_off + lm][0];
    const u32* br1 = (const u32*)&Ws[n_off + 16 + lm][0];
    for (int v = 0; v < 8; ++v) {
      af.u[v] = ar[hl * 8 + v];
      b0.u[v] = br0[hl * 8 + v];
      b1.u[v] = br1[hl * 8 + v];
    }
    c0.v = __builtin_amdgcn_wmma_f32_16x16x32_bf16(false, af.v, false, b0.v, (short)0, c0.v, false, false);
    c1.v = __builtin_amdgcn_wmma_f32_16x16x32_bf16(false, af.v, false, b1.v, (short)0, c1.v, false, false);
    __syncthreads();
  }
  for (int v = 0; v < 8; ++v) {
    int row = bm + m_off + v + 8 * hl;
    int col0 = bn + n_off + lm, col1 = col0 + 16;
    float o0 = c0.f[v] + bias[col0];
    float o1 = c1.f[v] + bias[col1];
    if (relu) { o0 = fmaxf(o0, 0.f); o1 = fmaxf(o1, 0.f); }
    C[(size_t)row * N + col0] = o0;
    C[(size_t)row * N + col1] = o1;
  }
}

// ---------------------------------------------------------------- flash attention (WMMA, bf16 inputs)
// Q,K: (B,T,NH,64) bf16 token-major; Vt: (B,NH,64,Tk) bf16.
__global__ __launch_bounds__(128) void attn_kernel(
    const u16* __restrict__ Q, const u16* __restrict__ K, const u16* __restrict__ Vt,
    float* __restrict__ O, int Tq, int Tk, float scale) {
  __shared__ __align__(16) u16 Ks[32][72];     // [j][d], 144B row stride
  __shared__ __align__(16) u16 Vs[64][40];     // [d][j]
  __shared__ __align__(16) u16 Ps[4][16][40];  // per-wave P [q][j]
  const int tid = threadIdx.x;
  const int lane = tid & 31, wid = tid >> 5;
  const int hl = lane >> 4, lm = lane & 15;
  const int b = blockIdx.y / NHh, h = blockIdx.y % NHh;
  const int q0 = blockIdx.x * 64 + wid * 16;
  const size_t bTk = (size_t)b * Tk;
  const size_t bh64 = ((size_t)b * NHh + h) * HDIM;

  FragB qa0, qa1;
  {
    const u32* qrow = (const u32*)(Q + ((size_t)b * Tq + q0 + lm) * DM + h * HDIM);
    for (int v = 0; v < 8; ++v) {
      int d2 = (v * 2 + hl * 16) >> 1;
      qa0.u[v] = qrow[d2];
      qa1.u[v] = qrow[d2 + 16];
    }
  }
  float mrow[8], lrow[8];
  AccF oacc[4];
  for (int v = 0; v < 8; ++v) { mrow[v] = -1e30f; lrow[v] = 0.f; }
  for (int t = 0; t < 4; ++t)
    for (int v = 0; v < 8; ++v) oacc[t].f[v] = 0.f;

  for (int jt = 0; jt < Tk; jt += 32) {
    for (int i = 0; i < 2; ++i) {
      int ci = tid + i * 128;
      int j = ci >> 3, c8 = ci & 7;
      cp16(K + (bTk + jt + j) * DM + h * HDIM + c8 * 8, &Ks[j][c8 * 8]);
      int d = ci >> 2, c4 = ci & 3;
      cp16(Vt + (bh64 + d) * Tk + jt + c4 * 8, &Vs[d][c4 * 8]);
    }
    if (jt + 32 < Tk)
      __builtin_prefetch(K + (bTk + jt + 32 + (tid >> 3)) * DM + h * HDIM + (tid & 7) * 8, 0, 1);
    async_wait();
    __syncthreads();
    AccF s0, s1;
    for (int v = 0; v < 8; ++v) { s0.f[v] = 0.f; s1.f[v] = 0.f; }
    {
      FragB kb;
      const u32* kr0 = (const u32*)&Ks[lm][0];
      const u32* kr1 = (const u32*)&Ks[16 + lm][0];
      for (int v = 0; v < 8; ++v) kb.u[v] = kr0[hl * 8 + v];
      s0.v = __builtin_amdgcn_wmma_f32_16x16x32_bf16(false, qa0.v, false, kb.v, (short)0, s0.v, false, false);
      for (int v = 0; v < 8; ++v) kb.u[v] = kr0[16 + hl * 8 + v];
      s0.v = __builtin_amdgcn_wmma_f32_16x16x32_bf16(false, qa1.v, false, kb.v, (short)0, s0.v, false, false);
      for (int v = 0; v < 8; ++v) kb.u[v] = kr1[hl * 8 + v];
      s1.v = __builtin_amdgcn_wmma_f32_16x16x32_bf16(false, qa0.v, false, kb.v, (short)0, s1.v, false, false);
      for (int v = 0; v < 8; ++v) kb.u[v] = kr1[16 + hl * 8 + v];
      s1.v = __builtin_amdgcn_wmma_f32_16x16x32_bf16(false, qa1.v, false, kb.v, (short)0, s1.v, false, false);
    }
    // online softmax over j (row m = v + 8*hl lives on lanes sharing bit4)
    for (int v = 0; v < 8; ++v) {
      float x0 = s0.f[v] * scale, x1 = s1.f[v] * scale;
      float mt = fmaxf(x0, x1);
      mt = fmaxf(mt, __shfl_xor(mt, 1));
      mt = fmaxf(mt, __shfl_xor(mt, 2));
      mt = fmaxf(mt, __shfl_xor(mt, 4));
      mt = fmaxf(mt, __shfl_xor(mt, 8));
      float mnew = fmaxf(mrow[v], mt);
      float alpha = __expf(mrow[v] - mnew);
      float p0 = __expf(x0 - mnew), p1 = __expf(x1 - mnew);
      float ps = p0 + p1;
      ps += __shfl_xor(ps, 1); ps += __shfl_xor(ps, 2);
      ps += __shfl_xor(ps, 4); ps += __shfl_xor(ps, 8);
      lrow[v] = lrow[v] * alpha + ps;
      mrow[v] = mnew;
      for (int t = 0; t < 4; ++t) oacc[t].f[v] *= alpha;
      int qr = v + 8 * hl;
      Ps[wid][qr][lm] = f2bf(p0);  // wave-private LDS; DS pipe is in-order per wave
      Ps[wid][qr][lm + 16] = f2bf(p1);
    }
    {
      FragB pa, vb;
      const u32* pr = (const u32*)&Ps[wid][lm][0];
      for (int v = 0; v < 8; ++v) pa.u[v] = pr[hl * 8 + v];
      for (int t = 0; t < 4; ++t) {
        const u32* vr = (const u32*)&Vs[t * 16 + lm][0];
        for (int v = 0; v < 8; ++v) vb.u[v] = vr[hl * 8 + v];
        oacc[t].v = __builtin_amdgcn_wmma_f32_16x16x32_bf16(false, pa.v, false, vb.v, (short)0, oacc[t].v, false, false);
      }
    }
    __syncthreads();
  }
  for (int t = 0; t < 4; ++t)
    for (int v = 0; v < 8; ++v) {
      int qg = q0 + v + 8 * hl;
      O[((size_t)b * Tq + qg) * DM + h * HDIM + t * 16 + lm] = oacc[t].f[v] / lrow[v];
    }
}

// ---------------------------------------------------------------- residual add + LayerNorm (rows of 256)
__global__ __launch_bounds__(256) void add_ln_kernel(float* __restrict__ x, const float* __restrict__ r,
                                                     const float* __restrict__ g, const float* __restrict__ bb) {
  __shared__ float red[256];
  size_t row = blockIdx.x;
  int t = threadIdx.x;
  float v = x[row * DM + t] + r[row * DM + t];
  red[t] = v;
  __syncthreads();
  for (int off = 128; off > 0; off >>= 1) {
    if (t < off) red[t] += red[t + off];
    __syncthreads();
  }
  float mean = red[0] * (1.0f / DM);
  __syncthreads();
  float d = v - mean;
  red[t] = d * d;
  __syncthreads();
  for (int off = 128; off > 0; off >>= 1) {
    if (t < off) red[t] += red[t + off];
    __syncthreads();
  }
  float var = red[0] * (1.0f / DM);
  x[row * DM + t] = d * rsqrtf(var + 1e-5f) * g[t] + bb[t];
}

__global__ void add_kernel(const float* __restrict__ a, const float* __restrict__ b,
                           float* __restrict__ c, int n) {
  int t = blockIdx.x * blockDim.x + threadIdx.x;
  if (t < n) c[t] = a[t] + b[t];
}

// ---------------------------------------------------------------- fourier positional encoding
__global__ void fourier_kernel(const float* __restrict__ xyz, const float* __restrict__ pcmin,
                               const float* __restrict__ pcmax, const float* __restrict__ gB,
                               float* __restrict__ out, int m) {
  int t = blockIdx.x * blockDim.x + threadIdx.x;
  if (t >= BATCH * m * 128) return;
  int b = t / (m * 128);
  int rem = t % (m * 128);
  int i = rem / 128, j = rem % 128;
  size_t row = (size_t)b * m + i;
  float s = 0.f;
  for (int c = 0; c < 3; ++c) {
    float xn = (xyz[row * 3 + c] - pcmin[b * 3 + c]) / (pcmax[b * 3 + c] - pcmin[b * 3 + c]);
    s += xn * gB[c * 128 + j];
  }
  s *= 2.0f * PI_F;
  out[row * DM + j] = __sinf(s);
  out[row * DM + 128 + j] = __cosf(s);
}

// ---------------------------------------------------------------- gather features by ball idx + sum
__global__ void gather_sum_kernel(const float* __restrict__ feat, const int* __restrict__ aidx,
                                  float* __restrict__ out) {
  int t = blockIdx.x * blockDim.x + threadIdx.x;
  if (t >= BATCH * NQc * DM) return;
  int d = t % DM;
  int q = (t / DM) % NQc;
  int b = t / (DM * NQc);
  float s = 0.f;
  const int* row = aidx + ((size_t)b * NQc + q) * NSAMP;
  for (int k = 0; k < NSAMP; ++k)
    s += feat[((size_t)b * NPRE + row[k]) * DM + d];
  out[((size_t)b * NQc + q) * DM + d] = s;
}

// ---------------------------------------------------------------- small GEMV for head l3
__global__ void gemv_kernel(const float* __restrict__ A, const float* __restrict__ w,
                            const float* __restrict__ bb, float* __restrict__ out, int rows, int n) {
  int t = blockIdx.x * blockDim.x + threadIdx.x;
  if (t >= rows * n) return;
  int row = t / n, o = t % n;
  float s = bb[o];
  const float* a = A + (size_t)row * DM;
  for (int k = 0; k < DM; ++k) s += a[k] * w[k * n + o];
  out[(size_t)row * n + o] = s;
}

// ---------------------------------------------------------------- final box assembly (74 outputs / row)
__global__ void final_kernel(const float* __restrict__ cls, const float* __restrict__ cen,
                             const float* __restrict__ siz, const float* __restrict__ acl,
                             const float* __restrict__ are, const float* __restrict__ qxyz,
                             const float* __restrict__ pcmin, const float* __restrict__ pcmax,
                             float* __restrict__ out) {
  const int R = NDECL * BATCH * NQc;
  int t = blockIdx.x * blockDim.x + threadIdx.x;
  if (t >= R) return;
  int q = t % NQc;
  int b = (t / NQc) % BATCH;
  const float* qx = qxyz + ((size_t)b * NQc + q) * 3;
  float ctr[3], sz[3];
  for (int c = 0; c < 3; ++c) {
    float sc = fmaxf(pcmax[b * 3 + c] - pcmin[b * 3 + c], 0.1f);
    ctr[c] = qx[c] + sigm(cen[(size_t)t * 3 + c]) - 0.5f;
    sz[c] = sigm(siz[(size_t)t * 3 + c]) * sc;
  }
  float ar[NANG];
  float best = -1e30f; int bi = 0;
  for (int k = 0; k < NANG; ++k) {
    float lg = acl[(size_t)t * NANG + k];
    ar[k] = are[(size_t)t * NANG + k] * (PI_F / NANG);
    if (lg > best) { best = lg; bi = k; }
  }
  float ang = (2.0f * PI_F / NANG) * bi + ar[bi];
  if (ang > PI_F) ang -= 2.0f * PI_F;
  float ca = __cosf(ang), sa = __sinf(ang);
  float* o = out + (size_t)t * 74;
  for (int c8 = 0; c8 < 8; ++c8) {
    float sx = (c8 & 4) ? 1.f : -1.f;
    float sy = (c8 & 2) ? 1.f : -1.f;
    float sz_ = (c8 & 1) ? 1.f : -1.f;
    float o0 = 0.5f * sz[0] * sx, o1 = 0.5f * sz[1] * sy, o2 = 0.5f * sz[2] * sz_;
    o[c8 * 3 + 0] = ca * o0 - sa * o1 + ctr[0];
    o[c8 * 3 + 1] = sa * o0 + ca * o1 + ctr[1];
    o[c8 * 3 + 2] = o2 + ctr[2];
  }
  for (int k = 0; k < NSEM + 1; ++k) o[24 + k] = cls[(size_t)t * (NSEM + 1) + k];
  for (int c = 0; c < 3; ++c) { o[43 + c] = ctr[c]; o[46 + c] = sz[c]; }
  for (int k = 0; k < NANG; ++k) { o[49 + k] = acl[(size_t)t * NANG + k]; o[61 + k] = ar[k]; }
  o[73] = ang;
}

// ================================================================ host
struct AttnP { const float *qw, *qb, *kw, *kb, *vw, *vb, *ow, *ob; };

extern "C" void kernel_launch(void* const* d_in, const int* in_sizes, int n_in,
                              void* d_out, int out_size, void* d_ws, size_t ws_size,
                              hipStream_t stream) {
  (void)in_sizes; (void)n_in; (void)out_size; (void)ws_size;
  int pi = 0;
  auto P = [&]() { return (const float*)d_in[pi++]; };
  auto PA = [&](AttnP& a) { a.qw=P();a.qb=P();a.kw=P();a.kb=P();a.vw=P();a.vb=P();a.ow=P();a.ob=P(); };

  const float* pc = P();  // point_clouds (B,N,3)
  const float *pm1w=P(),*pm1b=P(),*pm2w=P(),*pm2b=P(),*pm3w=P(),*pm3b=P();
  struct EncP { AttnP a; const float *f1w,*f1b,*f2w,*f2b,*g1,*bb1,*g2,*bb2; } enc[NENCL];
  for (int i = 0; i < NENCL; ++i) {
    PA(enc[i].a);
    enc[i].f1w=P();enc[i].f1b=P();enc[i].f2w=P();enc[i].f2b=P();
    enc[i].g1=P();enc[i].bb1=P();enc[i].g2=P();enc[i].bb2=P();
  }
  const float *pj1w=P(),*pj1b=P(),*pj2w=P(),*pj2b=P(),*pj3w=P(),*pj3b=P();
  const float *gaussB=P();
  const float *qp1w=P(),*qp1b=P(),*qp2w=P(),*qp2b=P();
  struct DecP { AttnP sa, ca; const float *f1w,*f1b,*f2w,*f2b,*g1,*bb1,*g2,*bb2,*g3,*bb3; } dec[NDECL];
  for (int i = 0; i < NDECL; ++i) {
    PA(dec[i].sa); PA(dec[i].ca);
    dec[i].f1w=P();dec[i].f1b=P();dec[i].f2w=P();dec[i].f2b=P();
    dec[i].g1=P();dec[i].bb1=P();dec[i].g2=P();dec[i].bb2=P();dec[i].g3=P();dec[i].bb3=P();
  }
  struct HeadP { const float *w1,*b1,*w2,*b2,*w3,*b3; } hd[5];
  for (int i = 0; i < 5; ++i) { hd[i].w1=P();hd[i].b1=P();hd[i].w2=P();hd[i].b2=P();hd[i].w3=P();hd[i].b3=P(); }

  // workspace carve
  char* wp = (char*)d_ws;
  auto alloc = [&](size_t bytes) { void* p = (void*)wp; wp += (bytes + 255) & ~(size_t)255; return p; };
  const size_t BIG = (size_t)BATCH * NPRE * DM;  // 2M floats
  const size_t SML = (size_t)BATCH * NQc * DM;   // 256K floats
  float* pcmin   = (float*)alloc(BATCH * 3 * 4);
  float* pcmax   = (float*)alloc(BATCH * 3 * 4);
  float* fpsdist = (float*)alloc((size_t)BATCH * NPTS * 4);
  int*   preind  = (int*)alloc((size_t)BATCH * NPRE * 4);
  float* prexyz  = (float*)alloc((size_t)BATCH * NPRE * 3 * 4);
  int*   nidx    = (int*)alloc((size_t)BATCH * NPRE * NSAMP * 4);
  float* x       = (float*)alloc(BIG * 4);
  float* tq      = (float*)alloc(BIG * 4);
  float* tk      = (float*)alloc(BIG * 4);
  float* tv      = (float*)alloc(BIG * 4);
  float* to      = (float*)alloc(BIG * 4);
  float* encfeat = (float*)alloc(BIG * 4);
  float* encpos  = (float*)alloc(BIG * 4);
  float* enckv   = (float*)alloc(BIG * 4);
  int*   qinds   = (int*)alloc((size_t)BATCH * NQc * 4);
  float* qxyz    = (float*)alloc((size_t)BATCH * NQc * 3 * 4);
  float* qpos    = (float*)alloc(SML * 4);
  float* qembed  = (float*)alloc(SML * 4);
  int*   aidx    = (int*)alloc((size_t)BATCH * NQc * NSAMP * 4);
  float* ybuf    = (float*)alloc(SML * 4);
  float* qin     = (float*)alloc(SML * 4);
  float* box     = (float*)alloc((size_t)NDECL * SML * 4);
  const int R = NDECL * BATCH * NQc;  // 8192 head rows
  float* h_cls = (float*)alloc((size_t)R * (NSEM + 1) * 4);
  float* h_cen = (float*)alloc((size_t)R * 3 * 4);
  float* h_siz = (float*)alloc((size_t)R * 3 * 4);
  float* h_acl = (float*)alloc((size_t)R * NANG * 4);
  float* h_are = (float*)alloc((size_t)R * NANG * 4);
  u16* abf  = (u16*)alloc(BIG * 2);              // bf16 activations for GEMM
  u16* wbf  = (u16*)alloc((size_t)DM * DM * 2);  // bf16 transposed weights
  u16* qbf  = (u16*)alloc(BIG * 2);              // bf16 Q for attention
  u16* kbf  = (u16*)alloc(BIG * 2);              // bf16 K
  u16* vtbf = (u16*)alloc(BIG * 2);              // bf16 V transposed (B,NH,64,Tk)

  auto GEMM = [&](const float* A, const float* w, const float* bb, float* C, int M, int relu) {
    int n = M * DM;
    cvt_bf16_kernel<<<(n + 255) / 256, 256, 0, stream>>>(A, abf, n);
    cvt_w_t_kernel<<<(DM * DM) / 256, 256, 0, stream>>>(w, wbf);
    gemm_bf_kernel<<<dim3(DM / 64, M / 64), 256, 0, stream>>>(abf, wbf, bb, C, M, DM, DM, relu);
  };
  auto ATTN = [&](const float* Qf, const float* Kf, const float* Vf, float* O, int Tq, int Tk) {
    cvt_bf16_kernel<<<(BATCH * Tq * DM + 255) / 256, 256, 0, stream>>>(Qf, qbf, BATCH * Tq * DM);
    cvt_bf16_kernel<<<(BATCH * Tk * DM + 255) / 256, 256, 0, stream>>>(Kf, kbf, BATCH * Tk * DM);
    cvt_v_t_kernel<<<(BATCH * NHh * HDIM * Tk + 255) / 256, 256, 0, stream>>>(Vf, vtbf, Tk);
    attn_kernel<<<dim3(Tq / 64, BATCH * NHh), 128, 0, stream>>>(qbf, kbf, vtbf, O, Tq, Tk, 0.125f);
  };
  const int MB = BATCH * NPRE;  // 8192 encoder tokens
  const int MQ = BATCH * NQc;   // 1024 decoder tokens

  // ---- preprocessing
  minmax_kernel<<<BATCH, 256, 0, stream>>>(pc, pcmin, pcmax);
  fps_kernel<<<BATCH, 512, 0, stream>>>(pc, preind, fpsdist, NPTS, NPRE);
  gather3_kernel<<<(BATCH * NPRE + 255) / 256, 256, 0, stream>>>(pc, preind, prexyz, NPTS, NPRE);
  ballq_kernel<<<(BATCH * NPRE + 255) / 256, 256, 0, stream>>>(pc, prexyz, nidx, NPTS, NPRE, 0.2f * 0.2f);
  premlp_kernel<<<BATCH * NPRE, 256, 0, stream>>>(pc, prexyz, nidx, pm1w, pm1b, pm2w, pm2b, pm3w, pm3b, x);

  // ---- encoder
  for (int l = 0; l < NENCL; ++l) {
    GEMM(x, enc[l].a.qw, enc[l].a.qb, tq, MB, 0);
    GEMM(x, enc[l].a.kw, enc[l].a.kb, tk, MB, 0);
    GEMM(x, enc[l].a.vw, enc[l].a.vb, tv, MB, 0);
    ATTN(tq, tk, tv, to, NPRE, NPRE);
    GEMM(to, enc[l].a.ow, enc[l].a.ob, tk, MB, 0);
    add_ln_kernel<<<MB, 256, 0, stream>>>(x, tk, enc[l].g1, enc[l].bb1);
    GEMM(x, enc[l].f1w, enc[l].f1b, tq, MB, 1);
    GEMM(tq, enc[l].f2w, enc[l].f2b, tk, MB, 0);
    add_ln_kernel<<<MB, 256, 0, stream>>>(x, tk, enc[l].g2, enc[l].bb2);
  }

  // ---- projection head
  GEMM(x, pj1w, pj1b, tq, MB, 1);
  GEMM(tq, pj2w, pj2b, tk, MB, 1);
  GEMM(tk, pj3w, pj3b, encfeat, MB, 1);

  // ---- queries
  fps_kernel<<<BATCH, 512, 0, stream>>>(prexyz, qinds, fpsdist, NPRE, NQc);
  gather3_kernel<<<(BATCH * NQc + 255) / 256, 256, 0, stream>>>(prexyz, qinds, qxyz, NPRE, NQc);
  fourier_kernel<<<(BATCH * NQc * 128 + 255) / 256, 256, 0, stream>>>(qxyz, pcmin, pcmax, gaussB, qpos, NQc);
  fourier_kernel<<<(BATCH * NPRE * 128 + 255) / 256, 256, 0, stream>>>(prexyz, pcmin, pcmax, gaussB, encpos, NPRE);
  GEMM(qpos, qp1w, qp1b, tq, MQ, 1);
  GEMM(tq, qp2w, qp2b, qembed, MQ, 1);
  ballq_kernel<<<(BATCH * NQc + 255) / 256, 256, 0, stream>>>(prexyz, qxyz, aidx, NPRE, NQc, 0.5f * 0.5f);
  gather_sum_kernel<<<(BATCH * NQc * DM + 255) / 256, 256, 0, stream>>>(encfeat, aidx, ybuf);
  add_kernel<<<(int)((BIG + 255) / 256), 256, 0, stream>>>(encfeat, encpos, enckv, (int)BIG);

  // ---- decoder
  for (int l = 0; l < NDECL; ++l) {
    add_kernel<<<(int)((SML + 255) / 256), 256, 0, stream>>>(ybuf, qembed, qin, (int)SML);
    GEMM(qin, dec[l].sa.qw, dec[l].sa.qb, tq, MQ, 0);
    GEMM(qin, dec[l].sa.kw, dec[l].sa.kb, tk, MQ, 0);
    GEMM(ybuf, dec[l].sa.vw, dec[l].sa.vb, tv, MQ, 0);
    ATTN(tq, tk, tv, to, NQc, NQc);
    GEMM(to, dec[l].sa.ow, dec[l].sa.ob, qin, MQ, 0);
    add_ln_kernel<<<MQ, 256, 0, stream>>>(ybuf, qin, dec[l].g1, dec[l].bb1);

    add_kernel<<<(int)((SML + 255) / 256), 256, 0, stream>>>(ybuf, qembed, qin, (int)SML);
    GEMM(qin, dec[l].ca.qw, dec[l].ca.qb, tq, MQ, 0);
    GEMM(enckv, dec[l].ca.kw, dec[l].ca.kb, tk, MB, 0);
    GEMM(encfeat, dec[l].ca.vw, dec[l].ca.vb, tv, MB, 0);
    ATTN(tq, tk, tv, to, NQc, NPRE);
    GEMM(to, dec[l].ca.ow, dec[l].ca.ob, qin, MQ, 0);
    add_ln_kernel<<<MQ, 256, 0, stream>>>(ybuf, qin, dec[l].g2, dec[l].bb2);

    GEMM(ybuf, dec[l].f1w, dec[l].f1b, tq, MQ, 1);
    GEMM(tq, dec[l].f2w, dec[l].f2b, to, MQ, 0);
    add_ln_kernel<<<MQ, 256, 0, stream>>>(ybuf, to, dec[l].g3, dec[l].bb3);

    (void)hipMemcpyAsync(box + (size_t)l * SML, ybuf, SML * sizeof(float),
                         hipMemcpyDeviceToDevice, stream);
  }

  // ---- heads
  float* outs[5] = {h_cls, h_cen, h_siz, h_acl, h_are};
  int odim[5] = {NSEM + 1, 3, 3, NANG, NANG};
  for (int i = 0; i < 5; ++i) {
    GEMM(box, hd[i].w1, hd[i].b1, tq, R, 1);
    GEMM(tq, hd[i].w2, hd[i].b2, tk, R, 1);
    gemv_kernel<<<(R * odim[i] + 255) / 256, 256, 0, stream>>>(tk, hd[i].w3, hd[i].b3, outs[i], R, odim[i]);
  }
  final_kernel<<<(R + 255) / 256, 256, 0, stream>>>(h_cls, h_cen, h_siz, h_acl, h_are,
                                                    qxyz, pcmin, pcmax, (float*)d_out);
}
